// TransformerBlock_52407190945812
// MI455X (gfx1250) — matrix operand order repaired and checked by reading, not compiled, not run on hardware
//
#include <hip/hip_runtime.h>
#include <hip/hip_bf16.h>

// ---------------- problem constants ----------------
static constexpr int K_B   = 2;
static constexpr int K_C   = 128;
static constexpr int K_H   = 192;
static constexpr int K_W   = 192;
static constexpr int K_HW  = K_H * K_W;        // 36864
static constexpr int K_NT  = K_B * K_HW;       // 73728
static constexpr int K_HID = 340;              // int(128*2.66)
static constexpr int K_HIDP= 344;              // padded to multiple of 8
static constexpr int K_ATT = 4608;             // 16^2+32^2+32^2+48^2

__constant__ int c_GS[4]     = {16, 32, 32, 48};
__constant__ int c_ATTOFF[4] = {0, 256, 1280, 2304};

static const int h_GS[4]     = {16, 32, 32, 48};
static const int h_GSTART[4] = {0, 16, 48, 80};
static const int h_ATTOFF[4] = {0, 256, 1280, 2304};

// ---------------- types ----------------
typedef __attribute__((ext_vector_type(16))) __bf16 v16bf;
typedef __attribute__((ext_vector_type(8)))  __bf16 v8bf;
typedef __attribute__((ext_vector_type(8)))  float  v8f;

__device__ __forceinline__ float gelu_f(float x) {
    return 0.5f * x * (1.0f + erff(x * 0.70710678118654752f));
}

// ---------------- WMMA fragment loaders (branch-free, 32-bit offsets) ----------------
// A fragment, row-major A[M,K], contiguous-K: two 16B vector loads.
__device__ __forceinline__ v16bf loadA_vec(const __bf16* __restrict__ A, int lda,
                                           int arow, int k0, int half) {
    const __bf16* p = A + (arow * lda + k0 + half * 8);
    v8bf a0 = *(const v8bf*)p;
    v8bf a1 = *(const v8bf*)(p + 16);
    v16bf r;
    #pragma unroll
    for (int e = 0; e < 8; ++e) { r[e] = a0[e]; r[e + 8] = a1[e]; }
    return r;
}
// B fragment, transB (B[N,K] row-major), contiguous-K: two 16B vector loads.
__device__ __forceinline__ v16bf loadB_T_vec(const __bf16* __restrict__ B, int ldb,
                                             int bcol, int k0, int half) {
    const __bf16* p = B + (bcol * ldb + k0 + half * 8);
    v8bf b0 = *(const v8bf*)p;
    v8bf b1 = *(const v8bf*)(p + 16);
    v16bf r;
    #pragma unroll
    for (int e = 0; e < 8; ++e) { r[e] = b0[e]; r[e + 8] = b1[e]; }
    return r;
}
// A fragment, transA (A[K,M]): K-major gather, no bounds checks (k in range).
__device__ __forceinline__ v16bf loadA_T_nc(const __bf16* __restrict__ A, int lda,
                                            int arow, int k0, int half) {
    v16bf r;
    #pragma unroll
    for (int e = 0; e < 16; ++e) {
        int kk = k0 + ((e < 8) ? (half * 8 + e) : (16 + half * 8 + (e - 8)));
        r[e] = A[kk * lda + arow];
    }
    return r;
}
// B fragment, normal (B[K,N]): K-major gather, no bounds checks.
__device__ __forceinline__ v16bf loadB_nc(const __bf16* __restrict__ B, int ldb,
                                          int bcol, int k0, int half) {
    v16bf r;
    #pragma unroll
    for (int e = 0; e < 16; ++e) {
        int kk = k0 + ((e < 8) ? (half * 8 + e) : (16 + half * 8 + (e - 8)));
        r[e] = B[kk * ldb + bcol];
    }
    return r;
}
// Guarded gathers for K tails: clamp index (always-legal load) + select zero.
__device__ __forceinline__ v16bf loadA_tail(const __bf16* __restrict__ A, int lda, int transA,
                                            int arow, int k0, int half, int kend, int Kmax) {
    v16bf r;
    #pragma unroll
    for (int e = 0; e < 16; ++e) {
        int kk = k0 + ((e < 8) ? (half * 8 + e) : (16 + half * 8 + (e - 8)));
        int kc = kk < Kmax ? kk : Kmax - 1;
        __bf16 v = transA ? A[kc * lda + arow] : A[arow * lda + kc];
        r[e] = kk < kend ? v : (__bf16)0.0f;
    }
    return r;
}
__device__ __forceinline__ v16bf loadB_tail(const __bf16* __restrict__ B, int ldb, int transB,
                                            int bcol, int k0, int half, int kend, int Kmax) {
    v16bf r;
    #pragma unroll
    for (int e = 0; e < 16; ++e) {
        int kk = k0 + ((e < 8) ? (half * 8 + e) : (16 + half * 8 + (e - 8)));
        int kc = kk < Kmax ? kk : Kmax - 1;
        __bf16 v = transB ? B[bcol * ldb + kc] : B[kc * ldb + bcol];
        r[e] = kk < kend ? v : (__bf16)0.0f;
    }
    return r;
}

// ---------------- generic WMMA GEMM ----------------
// C[M,N] = act( alpha*scaleA[m]*scaleB[n]*(A x B) + bias[m] ) (*emul) (+addsrc)
// transC: C stored pixel-major, ci = n*ldc + m (contiguous vector stores per lane).
// Fast path (!transA && transB && M%32==0): 32x16 output block per wave,
//   one B fragment feeds two WMMAs -> halves activation read traffic.
// splitK>1: K-slices accumulated into pre-zeroed Cf with atomicAdd.
__global__ void wmma_gemm_kernel(const __bf16* __restrict__ A, int lda, int transA,
                                 const __bf16* __restrict__ B, int ldb, int transB,
                                 float* __restrict__ Cf, __bf16* __restrict__ Cb,
                                 long ldc, int transC,
                                 int M, int N, int K, int splitK,
                                 const float* __restrict__ bias,
                                 const float* __restrict__ scaleA,
                                 const float* __restrict__ scaleB,
                                 const float* __restrict__ alpha_ptr,
                                 const float* __restrict__ addsrc,
                                 const __bf16* __restrict__ emul,
                                 int act)
{
    long wid   = (blockIdx.x * (long)blockDim.x + threadIdx.x) >> 5;
    int  lane  = threadIdx.x & 31;
    int  nsp   = splitK > 0 ? splitK : 1;
    long tile  = wid / nsp;
    int  slice = (int)(wid % nsp);
    bool dual  = (!transA && transB && (M & 31) == 0);
    int  mrows = dual ? 32 : 16;
    int tilesN = (N + 15) >> 4;
    int tm = (int)(tile / tilesN);
    int tn = (int)(tile % tilesN);
    if (tm * mrows >= M) return;               // wave-uniform
    int m0 = tm * mrows, n0 = tn * 16;
    int half = lane >> 4;
    int lr   = lane & 15;

    // clamped lanes feed only never-stored C rows/cols -> no zeroing needed
    int arow = min(m0 + lr, M - 1);
    int bcol = min(n0 + lr, N - 1);

    int kPer = ((K + nsp * 32 - 1) / (nsp * 32)) * 32;
    int kbeg = slice * kPer;
    int kend = min(kbeg + kPer, K);
    if (kbeg >= kend) return;                  // wave-uniform
    int kmain = kbeg + ((kend - kbeg) & ~31);

    v8f acc0 = {}, acc1 = {};
    if (dual) {                                // 32x16 block: B fragment reused twice
        int arow1 = m0 + 16 + lr;
        for (int k0 = kbeg; k0 < kmain; k0 += 32) {
            __builtin_prefetch(B + bcol * ldb + k0 + 128, 0, 1);
            v16bf vb  = loadB_T_vec(B, ldb, bcol, k0, half);
            v16bf va0 = loadA_vec(A, lda, arow, k0, half);
            acc0 = __builtin_amdgcn_wmma_f32_16x16x32_bf16(
                       false, va0, false, vb, (short)0, acc0, false, false);
            v16bf va1 = loadA_vec(A, lda, arow1, k0, half);
            acc1 = __builtin_amdgcn_wmma_f32_16x16x32_bf16(
                       false, va1, false, vb, (short)0, acc1, false, false);
        }
    } else if (!transA && transB) {            // 16x16, both contiguous-K vector loads
        for (int k0 = kbeg; k0 < kmain; k0 += 32) {
            __builtin_prefetch(B + bcol * ldb + k0 + 128, 0, 1);
            v16bf va = loadA_vec(A, lda, arow, k0, half);
            v16bf vb = loadB_T_vec(B, ldb, bcol, k0, half);
            acc0 = __builtin_amdgcn_wmma_f32_16x16x32_bf16(
                       false, va, false, vb, (short)0, acc0, false, false);
        }
    } else if (transA && !transB) {            // K-major gathers (attention / covariance)
        for (int k0 = kbeg; k0 < kmain; k0 += 32) {
            v16bf va = loadA_T_nc(A, lda, arow, k0, half);
            v16bf vb = loadB_nc(B, ldb, bcol, k0, half);
            acc0 = __builtin_amdgcn_wmma_f32_16x16x32_bf16(
                       false, va, false, vb, (short)0, acc0, false, false);
        }
    } else {                                   // generic (unused but safe)
        for (int k0 = kbeg; k0 < kmain; k0 += 32) {
            v16bf va = loadA_tail(A, lda, transA, arow, k0, half, kend, K);
            v16bf vb = loadB_tail(B, ldb, transB, bcol, k0, half, kend, K);
            acc0 = __builtin_amdgcn_wmma_f32_16x16x32_bf16(
                       false, va, false, vb, (short)0, acc0, false, false);
        }
    }
    if (kmain < kend) {                        // K tail (e.g. K=340), branch-free selects
        v16bf vb = loadB_tail(B, ldb, transB, bcol, kmain, half, kend, K);
        v16bf va = loadA_tail(A, lda, transA, arow, kmain, half, kend, K);
        acc0 = __builtin_amdgcn_wmma_f32_16x16x32_bf16(
                   false, va, false, vb, (short)0, acc0, false, false);
        if (dual) {
            v16bf va1 = loadA_tail(A, lda, transA, m0 + 16 + lr, kmain, half, kend, K);
            acc1 = __builtin_amdgcn_wmma_f32_16x16x32_bf16(
                       false, va1, false, vb, (short)0, acc1, false, false);
        }
    }

    // ---------------- epilogue ----------------
    float alpha = alpha_ptr ? *alpha_ptr : 1.0f;
    int cn = n0 + lr;
    if (cn >= N) return;                       // after all WMMAs
    float sB = scaleB ? scaleB[cn] : 1.0f;

    auto store_tile = [&](int mb, v8f accv) {
        if (nsp > 1) {                         // split-K partials: atomic f32 accumulate
            #pragma unroll
            for (int r = 0; r < 8; ++r) {
                int cm = mb + r + 8 * half;
                if (cm >= M) continue;
                float v = accv[r] * alpha * sB;
                if (scaleA) v *= scaleA[cm];
                long ci = transC ? (long)cn * ldc + cm : (long)cm * ldc + cn;
                atomicAdd(Cf + ci, v);
            }
            return;
        }
        float vals[8];
        bool fullM = (mb + 16) <= M;
        #pragma unroll
        for (int r = 0; r < 8; ++r) {
            int cm = mb + r + 8 * half;
            int cmc = min(cm, M - 1);
            float v = accv[r] * alpha * sB;
            if (scaleA) v *= scaleA[cmc];
            if (bias)   v += bias[cmc];
            if (act)    v  = gelu_f(v);
            long ci = transC ? (long)cn * ldc + cmc : (long)cmc * ldc + cn;
            if (emul)   v *= (float)emul[ci];
            if (addsrc) v += addsrc[ci];
            vals[r] = v;
        }
        if (transC && fullM) {                 // contiguous vector stores
            long base = (long)cn * ldc + mb + 8 * half;
            if (Cf) {
                float4 f0 = {vals[0], vals[1], vals[2], vals[3]};
                float4 f1 = {vals[4], vals[5], vals[6], vals[7]};
                *(float4*)(Cf + base)     = f0;
                *(float4*)(Cf + base + 4) = f1;
            }
            if (Cb) {
                v8bf h;
                #pragma unroll
                for (int r = 0; r < 8; ++r) h[r] = (__bf16)vals[r];
                *(v8bf*)(Cb + base) = h;
            }
        } else {
            #pragma unroll
            for (int r = 0; r < 8; ++r) {
                int cm = mb + r + 8 * half;
                if (cm >= M) continue;
                long ci = transC ? (long)cn * ldc + cm : (long)cm * ldc + cn;
                if (Cf) Cf[ci] = vals[r];
                if (Cb) Cb[ci] = (__bf16)vals[r];
            }
        }
    };
    store_tile(m0, acc0);
    if (dual) store_tile(m0 + 16, acc1);
}

// ---------------- small kernels ----------------
__global__ void zero_kernel(float* __restrict__ p, long n) {
    long i = blockIdx.x * (long)blockDim.x + threadIdx.x;
    if (i < n) p[i] = 0.f;
}

__global__ void cvt_f32_bf16_kernel(const float* __restrict__ s, __bf16* __restrict__ d, long n) {
    long i = blockIdx.x * (long)blockDim.x + threadIdx.x;
    if (i < n) d[i] = (__bf16)s[i];
}

// pad rows: d[r*ldd + c] = (c<cols) ? s[r*cols+c] : 0
__global__ void cvt_pad_kernel(const float* __restrict__ s, __bf16* __restrict__ d,
                               int rows, int cols, int ldd) {
    long i = blockIdx.x * (long)blockDim.x + threadIdx.x;
    if (i >= (long)rows * ldd) return;
    int r = (int)(i / ldd), c = (int)(i % ldd);
    d[i] = (__bf16)(c < cols ? s[(long)r * cols + c] : 0.f);
}

// LayerNorm over 128 channels, 32 pixels per block staged through LDS.
// bchw=1: in is (B,C,H,W) f32; bchw=0: in is pixel-major (NT,128) f32.
// out: pixel-major (NT,128) bf16, fully coalesced.
__global__ void ln_kernel(const float* __restrict__ in, const float* __restrict__ w,
                          const float* __restrict__ b, __bf16* __restrict__ out, int bchw)
{
    __shared__ float tile[128 * 32];           // [c*32 + nl]
    __shared__ float mu_s[32], iv_s[32];
    long pb = (long)blockIdx.x * 32;           // global pixel base
    int t = threadIdx.x;
    if (bchw) {
        int bb = (int)(pb / K_HW); int p0 = (int)(pb % K_HW);
        for (int i = t; i < 4096; i += 256) {
            int c = i / 32, nl = i % 32;       // pixels fastest -> coalesced
            tile[i] = in[((long)bb * K_C + c) * K_HW + p0 + nl];
        }
    } else {
        for (int i = t; i < 4096; i += 256) {
            int nl = i / 128, c = i % 128;     // channels fastest -> coalesced
            tile[c * 32 + nl] = in[(pb + nl) * 128 + c];
        }
    }
    __syncthreads();
    if (t < 32) {
        float mu = 0.f;
        for (int c = 0; c < 128; ++c) mu += tile[c * 32 + t];
        mu *= (1.0f / 128.0f);
        float var = 0.f;
        for (int c = 0; c < 128; ++c) { float d = tile[c * 32 + t] - mu; var += d * d; }
        var *= (1.0f / 128.0f);
        mu_s[t] = mu;
        iv_s[t] = rsqrtf(var + 1e-5f);
    }
    __syncthreads();
    for (int i = t; i < 4096; i += 256) {
        int nl = i / 128, c = i % 128;
        out[(pb + nl) * 128 + c] =
            (__bf16)((tile[c * 32 + nl] - mu_s[nl]) * iv_s[nl] * w[c] + b[c]);
    }
}

// depthwise 3x3 SAME. in/out pixel-major (NT, Ch) bf16, weights (Ch,1,3,3) f32
__global__ void dw3_kernel(const __bf16* __restrict__ in, const float* __restrict__ w,
                           __bf16* __restrict__ out, int Ch)
{
    long i = blockIdx.x * (long)blockDim.x + threadIdx.x;
    if (i >= (long)Ch * K_NT) return;
    int c = (int)(i % Ch);
    int n = (int)(i / Ch);
    int b = n / K_HW, p = n % K_HW;
    int y = p / K_W,  x = p % K_W;
    const float* wk = w + c * 9;
    const __bf16* base = in + ((long)b * K_HW) * Ch + c;
    float acc = 0.f;
    #pragma unroll
    for (int dy = -1; dy <= 1; ++dy) {
        int yy = y + dy;
        if (yy < 0 || yy >= K_H) continue;
        #pragma unroll
        for (int dx = -1; dx <= 1; ++dx) {
            int xx = x + dx;
            if (xx < 0 || xx >= K_W) continue;
            acc += wk[(dy + 1) * 3 + (dx + 1)] * (float)base[(long)(yy * K_W + xx) * Ch];
        }
    }
    out[(long)n * Ch + c] = (__bf16)acc;
}

// Coalesced column accumulation: 64 pixels/block, atomicAdd partials per 128 columns.
// sq: accumulate v*v; perbatch: separate accumulators per batch.
__global__ void colacc_kernel(const __bf16* __restrict__ m, long ld, int sq, int perbatch,
                              float* __restrict__ raw)
{
    long pbase = (long)blockIdx.x * 64;
    int t = threadIdx.x;
    int c = t & 127, sub = t >> 7;
    float acc = 0.f;
    for (int j = 0; j < 32; ++j) {
        long r = pbase + sub + 2 * j;
        float v = (float)m[r * ld + c];
        acc += sq ? v * v : v;
    }
    __shared__ float sm[256];
    sm[t] = acc; __syncthreads();
    if (t < 128) {
        int batch = perbatch ? (int)(pbase / K_HW) : 0;
        atomicAdd(raw + batch * 128 + c, sm[t] + sm[t + 128]);
    }
}

// stats[0:128] -> rowmean of cf ; stats[128:384]/[384:640] -> inv row norms of q/k
__global__ void finalize_rowmean_kernel(float* __restrict__ stats) {
    int i = threadIdx.x;
    if (i < 128) stats[i] *= (0.5f / (float)K_HW);
}
__global__ void finalize_inv_kernel(const float* __restrict__ stats,
                                    float* __restrict__ invq, float* __restrict__ invk) {
    int i = threadIdx.x;
    invq[i] = 1.0f / fmaxf(sqrtf(stats[128 + i]), 1e-12f);
    invk[i] = 1.0f / fmaxf(sqrtf(stats[384 + i]), 1e-12f);
}

// xm (HW,128) = cf - rowmean, bf16 (cf = batch mean of q channels of qkv_post)
__global__ void xm_kernel(const __bf16* __restrict__ qp, const float* __restrict__ rowmean,
                          __bf16* __restrict__ xm)
{
    long i = blockIdx.x * (long)blockDim.x + threadIdx.x;
    if (i >= (long)K_HW * 128) return;
    long l = i >> 7; int c = (int)(i & 127);
    float v = 0.5f * ((float)qp[l * 384 + c] + (float)qp[(l + K_HW) * 384 + c]);
    xm[i] = (__bf16)(v - rowmean[c]);
}

// stable descending argsort of corr row means (rank-based, 1 block / 128 threads)
__global__ void corr_argsort_kernel(const float* __restrict__ cov, int* __restrict__ idx) {
    __shared__ float dg[128], key[128];
    int i = threadIdx.x;
    dg[i] = sqrtf(cov[i * 128 + i]);
    __syncthreads();
    float s = 0.f;
    for (int j = 0; j < 128; ++j) s += cov[i * 128 + j] / (dg[i] * dg[j]);
    key[i] = s * (1.0f / 128.0f);
    __syncthreads();
    int r = 0;
    float ki = key[i];
    for (int j = 0; j < 128; ++j) {
        float kj = key[j];
        if (kj > ki || (kj == ki && j < i)) ++r;
    }
    idx[r] = i;
}

__global__ void permute_kernel(const __bf16* __restrict__ qkv, const int* __restrict__ idx,
                               __bf16* __restrict__ q_s, __bf16* __restrict__ k_s,
                               __bf16* __restrict__ v_s)
{
    long i = blockIdx.x * (long)blockDim.x + threadIdx.x;
    if (i >= (long)K_NT * 128) return;
    long n = i >> 7; int c = (int)(i & 127);
    int sc = idx[c];
    const __bf16* row = qkv + n * 384;
    q_s[i] = row[sc];
    k_s[i] = row[128 + sc];
    v_s[i] = row[256 + sc];
}

__global__ void softmax_kernel(float* __restrict__ att) {
    int gi = blockIdx.x, b = blockIdx.y;
    int g = c_GS[gi];
    float* a = att + b * K_ATT + c_ATTOFF[gi];
    int r = threadIdx.x;
    if (r >= g) return;
    float mx = -1e30f;
    for (int c = 0; c < g; ++c) mx = fmaxf(mx, a[r * g + c]);
    float s = 0.f;
    for (int c = 0; c < g; ++c) s += expf(a[r * g + c] - mx);
    float inv = 1.0f / s;
    for (int c = 0; c < g; ++c) a[r * g + c] = expf(a[r * g + c] - mx) * inv;
}

struct InterAll {
    const float* cw_w[4]; const float* cw_b[4];
    const float* g_w[4];  const float* g_b[4];
};

// inter block (pool-matrix math), one block per (group, batch), all in LDS
__global__ void inter_kernel(const float* __restrict__ qv, const float* __restrict__ att_in,
                             float* __restrict__ att_out_f, __bf16* __restrict__ att_out_b,
                             InterAll prm)
{
    int gi = blockIdx.x, b = blockIdx.y;
    int g = c_GS[gi], off = c_ATTOFF[gi];
    const float* attp = att_in + b * K_ATT + off;
    float*  outf = att_out_f + b * K_ATT + off;
    __bf16* outb = att_out_b + b * K_ATT + off;

    __shared__ float sm[13824];                // 55.3 KB of 320 KB WGP LDS
    float* s_att = sm;                         // 2304
    float* s_al  = sm + 2304;                  // 2304
    float* s_t   = sm + 4608;                  // 6144 (phase 1 only)
    float* s_y   = sm + 4608;                  // 4608 (reuses t region)
    float* s_x1  = sm + 9216;                  // 2304
    float* s_z   = sm + 11520;                 // 2304
    int tid = threadIdx.x;

    for (int i = tid; i < g * g; i += 256) s_att[i] = attp[i];
    for (int i = tid; i < g * 128; i += 256) {     // t = P qv
        int o = i / 128, j = i % 128;
        int so = o * 128 / g, eo = ((o + 1) * 128 + g - 1) / g;
        float a = 0.f;
        for (int q = so; q < eo; ++q) a += qv[q * 128 + j];
        s_t[i] = a / (float)(eo - so);
    }
    __syncthreads();
    for (int i = tid; i < g * g; i += 256) {       // aligned = t P^T
        int o = i / g, p = i % g;
        int sp = p * 128 / g, ep = ((p + 1) * 128 + g - 1) / g;
        float a = 0.f;
        for (int j = sp; j < ep; ++j) a += s_t[o * 128 + j];
        s_al[i] = a / (float)(ep - sp);
    }
    __syncthreads();
    const float* cw = prm.cw_w[gi]; const float* cb = prm.cw_b[gi];
    for (int i = tid; i < 2 * g * g; i += 256) {
        int r = i / g, p = i % g;
        float a = cb[r];
        for (int q = 0; q < g; ++q) a += cw[r * g + q] * (s_al[q * g + p] + s_att[q * g + p]);
        s_y[i] = a;
    }
    __syncthreads();
    for (int i = tid; i < g * g; i += 256) {
        int q = i / g, p = i % g;
        s_x1[i] = s_att[i] * s_y[q * g + p] + s_y[(g + q) * g + p];
    }
    __syncthreads();
    const float* gw = prm.g_w[gi]; const float* gb = prm.g_b[gi];
    for (int i = tid; i < g * g; i += 256) {
        int r = i / g, p = i % g;
        float a = gb[r];
        for (int q = 0; q < g; ++q) a += gw[r * g + q] * s_x1[q * g + p];
        s_z[i] = a;
    }
    __syncthreads();
    for (int i = tid; i < g * g; i += 256) {
        float v = s_x1[i] * gelu_f(s_z[i]) + s_att[i];
        outf[i] = v;
        outb[i] = (__bf16)v;
    }
}

__global__ void attmean_kernel(const float* __restrict__ att, float* __restrict__ amean) {
    int i = blockIdx.x * 256 + threadIdx.x;
    if (i < K_ATT) amean[i] = 0.5f * (att[i] + att[K_ATT + i]);
}

// bilinear upsample each group mean to 128x128, floor, sum; qv_new = 0.9*qv + 0.1*acc
__global__ void qvnew_kernel(const float* __restrict__ amean, const float* __restrict__ qv,
                             float* __restrict__ out)
{
    int i = blockIdx.x * 256 + threadIdx.x;
    if (i >= 128 * 128) return;
    int o1 = i / 128, o2 = i % 128;
    float acc = 0.f;
    for (int gi = 0; gi < 4; ++gi) {
        int g = c_GS[gi];
        const float* m = amean + c_ATTOFF[gi];
        float s  = (float)g / 128.0f;
        float sy = (o1 + 0.5f) * s - 0.5f;
        float sx = (o2 + 0.5f) * s - 0.5f;
        int y0 = (int)floorf(sy); float wy = sy - y0;
        int x0 = (int)floorf(sx); float wx = sx - x0;
        int y0c = min(max(y0, 0), g - 1), y1c = min(max(y0 + 1, 0), g - 1);
        int x0c = min(max(x0, 0), g - 1), x1c = min(max(x0 + 1, 0), g - 1);
        float v = m[y0c * g + x0c] * (1 - wy) * (1 - wx)
                + m[y0c * g + x1c] * (1 - wy) * wx
                + m[y1c * g + x0c] * wy * (1 - wx)
                + m[y1c * g + x1c] * wy * wx;
        acc += floorf(v);
    }
    out[i] = qv[i] * 0.9f + acc * 0.1f;
}

// s = qn + kn + out_all (all pixel-major) -> bf16
__global__ void s_kernel(const __bf16* __restrict__ q_s, const __bf16* __restrict__ k_s,
                         const float* __restrict__ invq, const float* __restrict__ invk,
                         const float* __restrict__ out_s, __bf16* __restrict__ s_b)
{
    long i = blockIdx.x * (long)blockDim.x + threadIdx.x;
    if (i >= (long)K_NT * 128) return;
    long n = i >> 7; int c = (int)(i & 127);
    int b = n >= K_HW;
    float v = (float)q_s[i] * invq[b * 128 + c] + (float)k_s[i] * invk[b * 128 + c] + out_s[i];
    s_b[i] = (__bf16)v;
}

// x1(pixel-major) = x(bchw) + proj(pixel-major), LDS-tiled transpose, both sides coalesced
__global__ void x1_kernel(const float* __restrict__ x, const float* __restrict__ pp,
                          float* __restrict__ x1)
{
    __shared__ float tile[32][33];             // [c_local][p_local]
    int p0 = blockIdx.x * 32, c0 = blockIdx.y * 32, b = blockIdx.z;
    int t = threadIdx.x, tr = t / 32, tc = t % 32;
    #pragma unroll
    for (int j = 0; j < 4; ++j) {
        int cl = tr + j * 8;
        tile[cl][tc] = x[((long)b * K_C + c0 + cl) * K_HW + p0 + tc];
    }
    __syncthreads();
    #pragma unroll
    for (int j = 0; j < 4; ++j) {
        int pl = tr + j * 8;
        long ci = ((long)b * K_HW + p0 + pl) * 128 + c0 + tc;
        x1[ci] = tile[tc][pl] + pp[ci];
    }
}

// out(bchw) = x1 + ffn_out (both pixel-major), LDS-tiled transpose
__global__ void final_kernel(const float* __restrict__ x1, const float* __restrict__ ff,
                             float* __restrict__ out)
{
    __shared__ float tile[32][33];             // [p_local][c_local]
    int p0 = blockIdx.x * 32, c0 = blockIdx.y * 32, b = blockIdx.z;
    int t = threadIdx.x, tr = t / 32, tc = t % 32;
    #pragma unroll
    for (int j = 0; j < 4; ++j) {
        int pl = tr + j * 8;
        long ci = ((long)b * K_HW + p0 + pl) * 128 + c0 + tc;
        tile[pl][tc] = x1[ci] + ff[ci];
    }
    __syncthreads();
    #pragma unroll
    for (int j = 0; j < 4; ++j) {
        int cl = tr + j * 8;
        out[((long)b * K_C + c0 + cl) * K_HW + p0 + tc] = tile[tc][cl];
    }
}

// h = gelu(y1)*y2, pixel-major; ypost (NT,680) -> h (NT,344 padded)
__global__ void h_kernel(const __bf16* __restrict__ ypost, __bf16* __restrict__ h) {
    long i = blockIdx.x * (long)blockDim.x + threadIdx.x;
    if (i >= (long)K_NT * K_HID) return;
    long n = i / K_HID; int c = (int)(i % K_HID);
    float y1 = (float)ypost[n * 680 + c];
    float y2 = (float)ypost[n * 680 + K_HID + c];
    h[n * K_HIDP + c] = (__bf16)(gelu_f(y1) * y2);
}

// ---------------- host-side helpers ----------------
static inline dim3 egrid(long n) { return dim3((unsigned)((n + 255) / 256)); }

static void launch_gemm(hipStream_t st,
                        const __bf16* A, int lda, int transA,
                        const __bf16* B, int ldb, int transB,
                        float* Cf, __bf16* Cb, long ldc, int transC,
                        int M, int N, int K, int splitK,
                        const float* bias, const float* sA, const float* sB,
                        const float* alpha, const float* addsrc, const __bf16* emul, int act)
{
    int nsp = splitK > 0 ? splitK : 1;
    int mrows = (!transA && transB && (M & 31) == 0) ? 32 : 16;   // must match kernel
    long waves  = (long)((M + mrows - 1) / mrows) * ((N + 15) / 16) * nsp;
    long blocks = (waves + 7) / 8;             // 8 wave32s per 256-thread block
    wmma_gemm_kernel<<<dim3((unsigned)blocks), dim3(256), 0, st>>>(
        A, lda, transA, B, ldb, transB, Cf, Cb, ldc, transC,
        M, N, K, nsp, bias, sA, sB, alpha, addsrc, emul, act);
}

extern "C" void kernel_launch(void* const* d_in, const int* in_sizes, int n_in,
                              void* d_out, int out_size, void* d_ws, size_t ws_size,
                              hipStream_t stream)
{
    (void)in_sizes; (void)n_in; (void)out_size; (void)ws_size;

    const float* x        = (const float*)d_in[0];
    const float* qv       = (const float*)d_in[1];
    const float* ln1_w    = (const float*)d_in[2];
    const float* ln1_b    = (const float*)d_in[3];
    const float* ln2_w    = (const float*)d_in[4];
    const float* ln2_b    = (const float*)d_in[5];
    const float* temp     = (const float*)d_in[6];
    const float* w_qkv    = (const float*)d_in[7];
    const float* w_qkv_dw = (const float*)d_in[8];
    const float* w_proj   = (const float*)d_in[9];
    const float* down_w   = (const float*)d_in[10];
    const float* down_b   = (const float*)d_in[11];
    const float* up_w     = (const float*)d_in[12];
    const float* up_b     = (const float*)d_in[13];
    const float* gate_w   = (const float*)d_in[14];
    const float* gate_b   = (const float*)d_in[15];
    const float* ffn_in_w = (const float*)d_in[32];
    const float* ffn_dw_w = (const float*)d_in[33];
    const float* ffn_out_w= (const float*)d_in[34];

    float* out    = (float*)d_out;
    float* out_qv = out + (size_t)K_B * K_C * K_HW;

    size_t off = 0;
    auto alloc = [&](size_t bytes) -> void* {
        off = (off + 255) & ~(size_t)255;
        void* p = (char*)d_ws + off;
        off += bytes;
        return p;
    };
    __bf16* xn       = (__bf16*)alloc((size_t)K_NT * 128 * 2);    // reused as xn2
    __bf16* wq_b     = (__bf16*)alloc((size_t)384 * 128 * 2);
    __bf16* wproj_b  = (__bf16*)alloc((size_t)128 * 128 * 2);
    __bf16* wgate_b  = (__bf16*)alloc((size_t)128 * 128 * 2);
    __bf16* wdown_b  = (__bf16*)alloc((size_t)64 * 128 * 2);
    __bf16* wup_b    = (__bf16*)alloc((size_t)128 * 64 * 2);
    __bf16* wfin_b   = (__bf16*)alloc((size_t)680 * 128 * 2);
    __bf16* wfout_b  = (__bf16*)alloc((size_t)128 * K_HIDP * 2);  // padded lda=344
    __bf16* qkv_pre  = (__bf16*)alloc((size_t)K_NT * 384 * 2);
    __bf16* qkv_post = (__bf16*)alloc((size_t)K_NT * 384 * 2);
    __bf16* xm       = (__bf16*)alloc((size_t)K_HW * 128 * 2);
    float*  cov      = (float*)alloc(128 * 128 * 4);
    float*  stats    = (float*)alloc(640 * 4);  // [rowmean|rawq|rawk]
    int*    idx      = (int*)alloc(128 * 4);
    float*  invq     = (float*)alloc(256 * 4);
    float*  invk     = (float*)alloc(256 * 4);
    __bf16* q_s      = (__bf16*)alloc((size_t)K_NT * 128 * 2);
    __bf16* k_s      = (__bf16*)alloc((size_t)K_NT * 128 * 2);
    __bf16* v_s      = (__bf16*)alloc((size_t)K_NT * 128 * 2);
    float*  att_f    = (float*)alloc((size_t)2 * K_ATT * 4);
    __bf16* att_b    = (__bf16*)alloc((size_t)2 * K_ATT * 2);
    float*  amean    = (float*)alloc((size_t)K_ATT * 4);
    float*  out_s    = (float*)alloc((size_t)K_NT * 128 * 4);
    __bf16* s_b      = (__bf16*)alloc((size_t)K_NT * 128 * 2);
    __bf16* gated    = (__bf16*)alloc((size_t)K_NT * 128 * 2);
    __bf16* dd       = (__bf16*)alloc((size_t)K_NT * 64 * 2);
    float*  out2_f   = (float*)alloc((size_t)K_NT * 128 * 4);
    __bf16* out2_b   = (__bf16*)alloc((size_t)K_NT * 128 * 2);
    float*  pp       = (float*)alloc((size_t)K_NT * 128 * 4);     // proj out / ffn out
    float*  x1       = (float*)alloc((size_t)K_NT * 128 * 4);
    __bf16* ffn_pre  = (__bf16*)alloc((size_t)K_NT * 680 * 2);
    __bf16* ypost    = (__bf16*)alloc((size_t)K_NT * 680 * 2);
    __bf16* hbuf     = (__bf16*)alloc((size_t)K_NT * K_HIDP * 2);

    auto cvt = [&](const float* s, __bf16* d, long n) {
        cvt_f32_bf16_kernel<<<egrid(n), 256, 0, stream>>>(s, d, n);
    };
    cvt(w_qkv, wq_b, 384 * 128);
    cvt(w_proj, wproj_b, 128 * 128);
    cvt(gate_w, wgate_b, 128 * 128);
    cvt(down_w, wdown_b, 64 * 128);
    cvt(up_w, wup_b, 128 * 64);
    cvt(ffn_in_w, wfin_b, 680 * 128);
    cvt_pad_kernel<<<egrid(128L * K_HIDP), 256, 0, stream>>>(ffn_out_w, wfout_b, 128, K_HID, K_HIDP);

    // LN1 -> xn (NT,128) bf16
    ln_kernel<<<dim3(K_NT / 32), 256, 0, stream>>>(x, ln1_w, ln1_b, xn, 1);
    // qkv = w_qkv @ xn, stored pixel-major (NT,384)
    launch_gemm(stream, wq_b, 128, 0, xn, 128, 1, nullptr, qkv_pre, 384, 1,
                384, K_NT, 128, 1, nullptr, nullptr, nullptr, nullptr, nullptr, nullptr, 0);
    dw3_kernel<<<egrid((long)384 * K_NT), 256, 0, stream>>>(qkv_pre, w_qkv_dw, qkv_post, 384);

    // correlation-based channel sort
    zero_kernel<<<egrid(640), 256, 0, stream>>>(stats, 640);
    zero_kernel<<<egrid(128 * 128), 256, 0, stream>>>(cov, 128 * 128);
    colacc_kernel<<<dim3(K_NT / 64), 256, 0, stream>>>(qkv_post, 384, 0, 0, stats);
    finalize_rowmean_kernel<<<1, 128, 0, stream>>>(stats);
    xm_kernel<<<egrid((long)K_HW * 128), 256, 0, stream>>>(qkv_post, stats, xm);
    launch_gemm(stream, xm, 128, 1, xm, 128, 0, cov, nullptr, 128, 0,
                128, 128, K_HW, 16, nullptr, nullptr, nullptr, nullptr, nullptr, nullptr, 0);
    corr_argsort_kernel<<<1, 128, 0, stream>>>(cov, idx);
    permute_kernel<<<egrid((long)K_NT * 128), 256, 0, stream>>>(qkv_post, idx, q_s, k_s, v_s);

    colacc_kernel<<<dim3(K_NT / 64), 256, 0, stream>>>(q_s, 128, 1, 1, stats + 128);
    colacc_kernel<<<dim3(K_NT / 64), 256, 0, stream>>>(k_s, 128, 1, 1, stats + 384);
    finalize_inv_kernel<<<1, 256, 0, stream>>>(stats, invq, invk);

    // att_raw = (qn kn^T)*temperature, split-K across hw with atomic accumulate
    zero_kernel<<<egrid(2 * K_ATT), 256, 0, stream>>>(att_f, 2 * K_ATT);
    for (int b = 0; b < K_B; ++b)
        for (int gi = 0; gi < 4; ++gi) {
            int g = h_GS[gi], st = h_GSTART[gi];
            const __bf16* qg = q_s + (size_t)b * K_HW * 128 + st;
            const __bf16* kg = k_s + (size_t)b * K_HW * 128 + st;
            launch_gemm(stream, qg, 128, 1, kg, 128, 0,
                        att_f + b * K_ATT + h_ATTOFF[gi], nullptr, g, 0,
                        g, g, K_HW, 64,
                        nullptr, invq + b * 128 + st, invk + b * 128 + st, temp + gi,
                        nullptr, nullptr, 0);
        }
    softmax_kernel<<<dim3(4, 2), 64, 0, stream>>>(att_f);

    InterAll ip;
    for (int i = 0; i < 4; ++i) {
        ip.cw_w[i] = (const float*)d_in[16 + 4 * i];
        ip.cw_b[i] = (const float*)d_in[17 + 4 * i];
        ip.g_w[i]  = (const float*)d_in[18 + 4 * i];
        ip.g_b[i]  = (const float*)d_in[19 + 4 * i];
    }
    inter_kernel<<<dim3(4, 2), 256, 0, stream>>>(qv, att_f, att_f, att_b, ip);

    // outs = att @ v, pixel-major output (transC)
    for (int b = 0; b < K_B; ++b)
        for (int gi = 0; gi < 4; ++gi) {
            int g = h_GS[gi], st = h_GSTART[gi];
            launch_gemm(stream, att_b + b * K_ATT + h_ATTOFF[gi], g, 0,
                        v_s + (size_t)b * K_HW * 128 + st, 128, 1,
                        out_s + (size_t)b * K_HW * 128 + st, nullptr, 128, 1,
                        g, K_HW, g, 1,
                        nullptr, nullptr, nullptr, nullptr, nullptr, nullptr, 0);
        }

    attmean_kernel<<<egrid(K_ATT), 256, 0, stream>>>(att_f, amean);
    qvnew_kernel<<<dim3(64), 256, 0, stream>>>(amean, qv, out_qv);

    // intra MLP (all pixel-major, fused epilogues)
    s_kernel<<<egrid((long)K_NT * 128), 256, 0, stream>>>(q_s, k_s, invq, invk, out_s, s_b);
    launch_gemm(stream, wgate_b, 128, 0, s_b, 128, 1, nullptr, gated, 128, 1,
                128, K_NT, 128, 1, gate_b, nullptr, nullptr, nullptr, nullptr, s_b, 1);
    launch_gemm(stream, wdown_b, 128, 0, gated, 128, 1, nullptr, dd, 64, 1,
                64, K_NT, 128, 1, down_b, nullptr, nullptr, nullptr, nullptr, nullptr, 0);
    launch_gemm(stream, wup_b, 64, 0, dd, 64, 1, out2_f, out2_b, 128, 1,
                128, K_NT, 64, 1, up_b, nullptr, nullptr, nullptr, out_s, nullptr, 0);
    launch_gemm(stream, wproj_b, 128, 0, out2_b, 128, 1, pp, nullptr, 128, 1,
                128, K_NT, 128, 1, nullptr, nullptr, nullptr, nullptr, nullptr, nullptr, 0);
    x1_kernel<<<dim3(K_HW / 32, 4, K_B), 256, 0, stream>>>(x, pp, x1);

    // FFN
    ln_kernel<<<dim3(K_NT / 32), 256, 0, stream>>>(x1, ln2_w, ln2_b, xn, 0);
    launch_gemm(stream, wfin_b, 128, 0, xn, 128, 1, nullptr, ffn_pre, 680, 1,
                680, K_NT, 128, 1, nullptr, nullptr, nullptr, nullptr, nullptr, nullptr, 0);
    dw3_kernel<<<egrid((long)680 * K_NT), 256, 0, stream>>>(ffn_pre, ffn_dw_w, ypost, 680);
    h_kernel<<<egrid((long)K_NT * K_HID), 256, 0, stream>>>(ypost, hbuf);
    launch_gemm(stream, wfout_b, K_HIDP, 0, hbuf, K_HIDP, 1, pp, nullptr, 128, 1,
                128, K_NT, K_HID, 1, nullptr, nullptr, nullptr, nullptr, nullptr, nullptr, 0);
    final_kernel<<<dim3(K_HW / 32, 4, K_B), 256, 0, stream>>>(x1, pp, out);
}
